// ANI_66357244723536
// MI455X (gfx1250) — compile-verified, matrix operand order checked by reading, not run on hardware
//
#include <hip/hip_runtime.h>
#include <hip/hip_bf16.h>

// ---- problem constants (match reference) ----
#define BB   32
#define AA   1024
#define DD   384
#define EE   4
#define H1   256
#define H2   192
#define ATOMS (BB*AA)          // 32768
#define WGATOMS 64
#define NWG   (ATOMS/WGATOMS)  // 512 workgroups

// tiles
#define KT1  (DD/32)           // 12
#define NT1  (H1/16)           // 16
#define KT2  (H1/32)           // 8
#define NT2  (H2/16)           // 12

typedef __attribute__((ext_vector_type(16))) __bf16 v16bf;
typedef __attribute__((ext_vector_type(8)))  __bf16 v8bf;
typedef __attribute__((ext_vector_type(8)))  float  v8f;

// fragment buffer sizes (bf16 elements): each frag = 32 lanes * 16 vals = 512
#define W1FRAGS (EE*KT1*NT1)           // 768 frags
#define WHFRAGS (EE*KT2*NT2)           // 384 frags
#define W1F_ELEMS (W1FRAGS*512)        // 393216
#define WHF_ELEMS (WHFRAGS*512)        // 196608

// ------------------------------------------------------------------
__global__ void ani_zero_out(float* out) {
    if (threadIdx.x < BB) out[threadIdx.x] = 0.0f;
}

// ------------------------------------------------------------------
// Repack W1 [E][384][256] f32 -> bf16 WMMA B-fragments.
// B-frag (16-bit, 32x16): lane l holds column n = (l&15); khalf=(l>>4)*8;
// vals j=0..7 -> K=khalf+j ; j=8..15 -> K=16+khalf+(j-8)
__global__ void ani_pack_w1(const float* __restrict__ W1, __bf16* __restrict__ dst) {
    int tid  = blockIdx.x * blockDim.x + threadIdx.x;
    if (tid >= W1FRAGS * 32) return;
    int lane = tid & 31;
    int f    = tid >> 5;
    int nt   = f % NT1;
    int kt   = (f / NT1) % KT1;
    int e    = f / (NT1 * KT1);
    int n     = nt * 16 + (lane & 15);
    int khalf = (lane >> 4) * 8;
    __bf16* o = dst + (size_t)f * 512 + lane * 16;
#pragma unroll
    for (int j = 0; j < 16; ++j) {
        int kl = (j < 8) ? (khalf + j) : (16 + khalf + (j - 8));
        int k  = kt * 32 + kl;
        o[j] = (__bf16)W1[((size_t)e * DD + k) * H1 + n];
    }
}

__global__ void ani_pack_wh(const float* __restrict__ Wh, __bf16* __restrict__ dst) {
    int tid  = blockIdx.x * blockDim.x + threadIdx.x;
    if (tid >= WHFRAGS * 32) return;
    int lane = tid & 31;
    int f    = tid >> 5;
    int nt   = f % NT2;
    int kt   = (f / NT2) % KT2;
    int e    = f / (NT2 * KT2);
    int n     = nt * 16 + (lane & 15);
    int khalf = (lane >> 4) * 8;
    __bf16* o = dst + (size_t)f * 512 + lane * 16;
#pragma unroll
    for (int j = 0; j < 16; ++j) {
        int kl = (j < 8) ? (khalf + j) : (16 + khalf + (j - 8));
        int k  = kt * 32 + kl;
        o[j] = (__bf16)Wh[((size_t)e * H1 + k) * H2 + n];
    }
}

// ------------------------------------------------------------------
// Main fused kernel: 64 atoms per workgroup, 16 waves.
// wave = (e = wave>>2, ng = wave&3):
//   layer1: 4 M-tiles x 4 N-tiles per wave  (N range ng*64..ng*64+63)
//   layer2: 4 M-tiles x 3 N-tiles per wave  (N range ng*48..ng*48+47)
__global__ __launch_bounds__(512)
void ani_mlp_kernel(const float* __restrict__ rep,
                    const int*   __restrict__ species,
                    const __bf16* __restrict__ w1f,
                    const __bf16* __restrict__ whf,
                    const float* __restrict__ b1,
                    const float* __restrict__ bh,
                    const float* __restrict__ W2,
                    const float* __restrict__ b2,
                    float* __restrict__ out) {
    __shared__ __bf16 lds_x[WGATOMS * DD];        // 48 KB
    __shared__ __bf16 lds_h1[EE][WGATOMS * H1];   // 128 KB
    __shared__ float  lds_red[16];

    const int tid   = threadIdx.x;
    const int wave  = tid >> 5;
    const int lane  = tid & 31;
    const int e     = wave >> 2;
    const int ng    = wave & 3;
    const int atom0 = blockIdx.x * WGATOMS;

    // ---- stage 64x384 x-tile to LDS as bf16 ----
    for (int idx = tid; idx < WGATOMS * DD; idx += 512) {
        lds_x[idx] = (__bf16)rep[(size_t)atom0 * DD + idx];
    }
    __syncthreads();

    const int nlo   = lane & 15;
    const int khalf = (lane >> 4) * 8;
    const int mhalf = (lane >> 4) * 8;

    // =========== layer 1: [64x384] x [384x256] + b1 (no activation) ======
    v8f acc[4][4];   // [mt][nt]
#pragma unroll
    for (int nt = 0; nt < 4; ++nt) {
        float bv = b1[e * H1 + (ng * 4 + nt) * 16 + nlo];
#pragma unroll
        for (int mt = 0; mt < 4; ++mt)
#pragma unroll
            for (int j = 0; j < 8; ++j) acc[mt][nt][j] = bv;
    }

    for (int kt = 0; kt < KT1; ++kt) {
        union { v16bf v; v8bf h[2]; } A[4];
#pragma unroll
        for (int mt = 0; mt < 4; ++mt) {
            const __bf16* xp = &lds_x[(mt * 16 + nlo) * DD + kt * 32 + khalf];
            A[mt].h[0] = *(const v8bf*)(xp);
            A[mt].h[1] = *(const v8bf*)(xp + 16);
        }
        const __bf16* wp = w1f
            + (size_t)((e * KT1 + kt) * NT1 + ng * 4) * 512 + lane * 16;
        v16bf Bf[4];
#pragma unroll
        for (int nt = 0; nt < 4; ++nt)
            Bf[nt] = *(const v16bf*)(wp + (size_t)nt * 512);
#pragma unroll
        for (int nt = 0; nt < 4; ++nt)
#pragma unroll
            for (int mt = 0; mt < 4; ++mt)
                acc[mt][nt] = __builtin_amdgcn_wmma_f32_16x16x32_bf16(
                    false, A[mt].v, false, Bf[nt], (short)0, acc[mt][nt],
                    false, false);
    }

    // ---- write h1 (bf16) into per-element LDS panel ----
    {
        __bf16* hp = &lds_h1[e][0];
#pragma unroll
        for (int mt = 0; mt < 4; ++mt)
#pragma unroll
            for (int nt = 0; nt < 4; ++nt) {
                int n = (ng * 4 + nt) * 16 + nlo;
#pragma unroll
                for (int i = 0; i < 8; ++i)
                    hp[(mt * 16 + mhalf + i) * H1 + n] = (__bf16)acc[mt][nt][i];
            }
    }
    __syncthreads();

    // =========== layer 2: [64x256] x [256x192] + bh (ReLU deferred) ======
    v8f acc2[4][3];  // [mt][ntl], global nt = ng*3 + ntl
#pragma unroll
    for (int ntl = 0; ntl < 3; ++ntl) {
        float bv = bh[e * H2 + (ng * 3 + ntl) * 16 + nlo];
#pragma unroll
        for (int mt = 0; mt < 4; ++mt)
#pragma unroll
            for (int j = 0; j < 8; ++j) acc2[mt][ntl][j] = bv;
    }

    for (int kt = 0; kt < KT2; ++kt) {
        union { v16bf v; v8bf h[2]; } A[4];
#pragma unroll
        for (int mt = 0; mt < 4; ++mt) {
            const __bf16* hq = &lds_h1[e][(mt * 16 + nlo) * H1 + kt * 32 + khalf];
            A[mt].h[0] = *(const v8bf*)(hq);
            A[mt].h[1] = *(const v8bf*)(hq + 16);
        }
        const __bf16* wq = whf
            + (size_t)((e * KT2 + kt) * NT2 + ng * 3) * 512 + lane * 16;
        v16bf Bf[3];
#pragma unroll
        for (int ntl = 0; ntl < 3; ++ntl)
            Bf[ntl] = *(const v16bf*)(wq + (size_t)ntl * 512);
#pragma unroll
        for (int ntl = 0; ntl < 3; ++ntl)
#pragma unroll
            for (int mt = 0; mt < 4; ++mt)
                acc2[mt][ntl] = __builtin_amdgcn_wmma_f32_16x16x32_bf16(
                    false, A[mt].v, false, Bf[ntl], (short)0, acc2[mt][ntl],
                    false, false);
    }

    // =========== layer 3: relu(h2) . W2[e] + b2[e], species mask =========
    float w2v[3];
#pragma unroll
    for (int ntl = 0; ntl < 3; ++ntl)
        w2v[ntl] = W2[e * H2 + (ng * 3 + ntl) * 16 + nlo];

    float part = 0.0f;
#pragma unroll
    for (int mt = 0; mt < 4; ++mt) {
#pragma unroll
        for (int i = 0; i < 8; ++i) {
            int m  = mt * 16 + mhalf + i;
            int sp = species[atom0 + m];
            if (sp == e) {
                float s = (ng == 0 && nlo == 0) ? b2[e] : 0.0f;
#pragma unroll
                for (int ntl = 0; ntl < 3; ++ntl) {
                    float h = acc2[mt][ntl][i];
                    h = h > 0.0f ? h : 0.0f;
                    s += h * w2v[ntl];
                }
                part += s;
            }
        }
    }

    // wave32 reduction, then one atomic per workgroup
#pragma unroll
    for (int off = 16; off > 0; off >>= 1)
        part += __shfl_xor(part, off, 32);
    if (lane == 0) lds_red[wave] = part;
    __syncthreads();
    if (tid == 0) {
        float tot = 0.0f;
#pragma unroll
        for (int w = 0; w < 16; ++w) tot += lds_red[w];
        atomicAdd(&out[atom0 / AA], tot);
    }
}

// ------------------------------------------------------------------
extern "C" void kernel_launch(void* const* d_in, const int* in_sizes, int n_in,
                              void* d_out, int out_size, void* d_ws, size_t ws_size,
                              hipStream_t stream) {
    const float* rep     = (const float*)d_in[0];
    const int*   species = (const int*)  d_in[1];
    const float* W1      = (const float*)d_in[2];
    const float* b1      = (const float*)d_in[3];
    const float* Wh      = (const float*)d_in[4];
    const float* bh      = (const float*)d_in[5];
    const float* W2      = (const float*)d_in[6];
    const float* b2      = (const float*)d_in[7];
    float* out = (float*)d_out;

    __bf16* w1f = (__bf16*)d_ws;                       // 768 KB
    __bf16* whf = (__bf16*)d_ws + W1F_ELEMS;           // +384 KB

    ani_zero_out<<<1, 32, 0, stream>>>(out);
    ani_pack_w1<<<(W1FRAGS * 32 + 255) / 256, 256, 0, stream>>>(W1, w1f);
    ani_pack_wh<<<(WHFRAGS * 32 + 255) / 256, 256, 0, stream>>>(Wh, whf);
    ani_mlp_kernel<<<NWG, 512, 0, stream>>>(rep, species, w1f, whf,
                                            b1, bh, W2, b2, out);
}
